// MultiHeadAttention_87359634801158
// MI455X (gfx1250) — matrix operand order repaired and checked by reading, not compiled, not run on hardware
//
#include <hip/hip_runtime.h>

#define HID 1024
#define NHEADS 16
#define HD 64
#define BATCH 2
#define SEQ 2048

typedef __attribute__((ext_vector_type(16))) _Float16 v16h;
typedef __attribute__((ext_vector_type(8)))  _Float16 v8h;
typedef __attribute__((ext_vector_type(8)))  float    v8f;
typedef __attribute__((ext_vector_type(4))) unsigned int u32x4;
typedef __attribute__((ext_vector_type(8))) int i32x8;
typedef __attribute__((ext_vector_type(4))) int i32x4;

#if defined(__gfx1250__) && __has_builtin(__builtin_amdgcn_tensor_load_to_lds)
#define HAVE_TDM 1
#else
#define HAVE_TDM 0
#endif

static __device__ __forceinline__ v16h join8(v8h a, v8h b) {
  return __builtin_shufflevector(a, b, 0,1,2,3,4,5,6,7,8,9,10,11,12,13,14,15);
}

// 2D f16 tile (TILE_W elems/row, TILE_H rows, row stride STRIDE elems) ->
// contiguous LDS, via the Tensor Data Mover (ISA ch.8 D# layout).
// 6-operand builtin: (u32x4 g0, i32x8 g1, i32x4 g2, i32x4 g3, i32x8, i32 cpol)
template <int TILE_W, int TILE_H, int STRIDE_ELT>
static __device__ __forceinline__ void tdm_load_2d_f16(unsigned lds_byte_off,
                                                       const _Float16* gsrc) {
#if HAVE_TDM
  unsigned long long ga = (unsigned long long)(size_t)gsrc;
  u32x4 g0 = {1u,                                   // count=1 (valid, user)
              lds_byte_off,                         // lds_addr (bytes)
              (unsigned)(ga & 0xFFFFFFFFu),         // global_addr[31:0]
              (unsigned)((ga >> 32) & 0x1FFFFFFu) | (2u << 30)};  // type=2
  i32x8 g1 = {0x00010000,                           // data_size=1 (2 bytes)
              (int)((unsigned)TILE_W << 16),        // tensor_dim0[15:0]
              (int)((unsigned)TILE_H << 16),        // dim0 hi | tensor_dim1 lo
              (int)((unsigned)TILE_W << 16),        // dim1 hi | tile_dim0
              (int)TILE_H,                          // tile_dim1 (tile_dim2=0)
              (int)STRIDE_ELT,                      // tensor_dim0_stride lo32
              0, 0};
  i32x4 gz4 = {0, 0, 0, 0};
  i32x8 gz8 = {0, 0, 0, 0, 0, 0, 0, 0};
  __builtin_amdgcn_tensor_load_to_lds(g0, g1, gz4, gz4, gz8, 0);
#else
  (void)lds_byte_off; (void)gsrc;
#endif
}

// ---------------------------------------------------------------------------
// C[m,n] = sum_k A[m,k] * W[n,k] + bias[n];  M = BATCH*SEQ, N = K = HID.
// Block: 128 threads (4 waves), 64(M) x 64(N) tile, K-step 32.
// ---------------------------------------------------------------------------
template <typename AT>
__global__ __launch_bounds__(128) void gemm_rowmajor_wt(
    const AT* __restrict__ A, const float* __restrict__ W,
    const float* __restrict__ bias,
    float* __restrict__ Cf, _Float16* __restrict__ Ch) {
  __shared__ __attribute__((aligned(32))) _Float16 As[64][32];
  __shared__ __attribute__((aligned(32))) _Float16 Bs[64][32];
  const int tid  = threadIdx.x;
  const int wave = tid >> 5, lane = tid & 31;
  const int hi = lane >> 4, ln = lane & 15;
  const int nblk = HID / 64;
  const int m0 = (int)(blockIdx.x / nblk) * 64;
  const int n0 = (int)(blockIdx.x % nblk) * 64;

  v8f acc[4] = {};
  const int lrow = tid >> 1, lcol = (tid & 1) << 4;

  for (int k0 = 0; k0 < HID; k0 += 32) {
    __syncthreads();
    const AT*    asrc = A + (size_t)(m0 + lrow) * HID + k0 + lcol;
    const float* wsrc = W + (size_t)(n0 + lrow) * HID + k0 + lcol;
#pragma unroll
    for (int j = 0; j < 16; ++j) {
      As[lrow][lcol + j] = (_Float16)(float)asrc[j];
      Bs[lrow][lcol + j] = (_Float16)wsrc[j];
    }
    __syncthreads();

    const _Float16* arow = &As[wave * 16 + ln][0];
    v16h af = join8(*(const v8h*)(arow + hi * 8),
                    *(const v8h*)(arow + 16 + hi * 8));
#pragma unroll
    for (int t = 0; t < 4; ++t) {
      v16h bfrag = join8(*(const v8h*)&Bs[t * 16 + ln][hi * 8],
                         *(const v8h*)&Bs[t * 16 + ln][16 + hi * 8]);
      acc[t] = __builtin_amdgcn_wmma_f32_16x16x32_f16(
          false, af, false, bfrag, (short)0, acc[t], false, false);
    }
  }

#pragma unroll
  for (int t = 0; t < 4; ++t) {
#pragma unroll
    for (int r = 0; r < 8; ++r) {
      int m = m0 + wave * 16 + hi * 8 + r;
      int n = n0 + t * 16 + ln;
      float v = acc[t][r] + bias[n];
      if (Cf) Cf[(size_t)m * HID + n] = v;
      if (Ch) Ch[(size_t)m * HID + n] = (_Float16)v;
    }
  }
}

// ---------------------------------------------------------------------------
// Relative-position attention. Block = one (b,h) x 128 q-rows; 8 waves x 16 q.
// Two-pass softmax. K tiles double-buffered through the Tensor Data Mover:
// tile i+1 is in flight (TENSORcnt=1) while tile i is consumed by WMMAs.
// ---------------------------------------------------------------------------
__global__ __launch_bounds__(256) void relattn_kernel(
    const _Float16* __restrict__ qh, const _Float16* __restrict__ kh,
    const _Float16* __restrict__ vh,
    const float* __restrict__ relk, const float* __restrict__ relv,
    _Float16* __restrict__ ctx) {
  constexpr int QT = 128, KT = 64;
  __shared__ __attribute__((aligned(32))) _Float16 Ks[2][KT][HD]; // [key][d]
  __shared__ __attribute__((aligned(32))) _Float16 Vt[HD][KT];    // [d][key]
  __shared__ __attribute__((aligned(32))) _Float16 Ps[8][16][KT];
  __shared__ float qr_s[QT][8];   // 7 used
  __shared__ float cen_s[QT][6];  // 5 used

  const int tid = threadIdx.x;
  const int wave = tid >> 5, lane = tid & 31;
  const int hi = lane >> 4, ln = lane & 15;
  const int nqt = SEQ / QT;
  const int bh = (int)blockIdx.x / nqt;
  const int q0 = ((int)blockIdx.x % nqt) * QT;
  const int b = bh / NHEADS, h = bh % NHEADS;
  const size_t base = (size_t)b * SEQ * HID + (size_t)h * HD;
  const int qw0 = q0 + wave * 16;     // first q row of this wave

  for (int i = tid; i < QT * 6; i += 256) ((float*)cen_s)[i] = 0.f;

  // qr[q][r] = q-row . rel_k_table[r]
  for (int i = tid; i < QT * 7; i += 256) {
    int ql = i / 7, r = i % 7;
    const _Float16* qrow = qh + base + (size_t)(q0 + ql) * HID;
    float acc = 0.f;
    for (int d = 0; d < HD; ++d) acc += (float)qrow[d] * relk[r * HD + d];
    qr_s[ql][r] = acc;
  }

  // Q A-fragments (d = 0..31 / 32..63), ISA 16-bit A layout.
  v16h a0, a1;
  {
    const _Float16* qrow = qh + base + (size_t)(qw0 + ln) * HID;
#pragma unroll
    for (int i = 0; i < 16; ++i) {
      int koff = ((i >> 3) << 4) + hi * 8 + (i & 7);
      a0[i] = qrow[koff];
      a1[i] = qrow[32 + koff];
    }
  }
  __syncthreads();

  // pre-scaled tail biases per row (bucket 0 = far left, bucket 6 = far right)
  float bz0[8], bz6[8];
#pragma unroll
  for (int r = 0; r < 8; ++r) {
    int ql = wave * 16 + hi * 8 + r;
    bz0[r] = qr_s[ql][0] * 0.125f;
    bz6[r] = qr_s[ql][6] * 0.125f;
  }

  float mr[8], lr[8];
#pragma unroll
  for (int r = 0; r < 8; ++r) { mr[r] = -1e30f; lr[r] = 0.f; }

  // ---------------- pass 1: exact row max / logsumexp ----------------
#if HAVE_TDM
  if (tid < 32)   // prologue: tile 0 into buffer 0
    tdm_load_2d_f16<HD, KT, HID>((unsigned)(size_t)&Ks[0][0][0], kh + base);
#endif
  for (int kt = 0; kt < SEQ; kt += KT) {
    const int cur = (kt / KT) & 1;
    __syncthreads();   // readers of buffer cur^1 (iter i-1) are done
#if HAVE_TDM
    if (tid < 32) {
      if (kt + KT < SEQ) {     // launch next tile into the other buffer
        tdm_load_2d_f16<HD, KT, HID>((unsigned)(size_t)&Ks[cur ^ 1][0][0],
                                     kh + base + (size_t)(kt + KT) * HID);
        __builtin_amdgcn_s_wait_tensorcnt(1);   // current tile complete
      } else {
        __builtin_amdgcn_s_wait_tensorcnt(0);
      }
    }
#else
    for (int i = tid; i < KT * HD / 8; i += 256) {
      int key = i >> 3, dc = (i & 7) << 3;
      *(v8h*)&Ks[cur][key][dc] =
          *(const v8h*)(kh + base + (size_t)(kt + key) * HID + dc);
    }
#endif
    __syncthreads();

    const bool leftU  = (kt + KT - 1) <= (qw0 - 4);
    const bool rightU = kt >= (qw0 + 18);
#pragma unroll
    for (int t = 0; t < 4; ++t) {
      v8f c = {};
      v16h b0 = join8(*(const v8h*)&Ks[cur][t * 16 + ln][hi * 8],
                      *(const v8h*)&Ks[cur][t * 16 + ln][16 + hi * 8]);
      v16h b1 = join8(*(const v8h*)&Ks[cur][t * 16 + ln][32 + hi * 8],
                      *(const v8h*)&Ks[cur][t * 16 + ln][48 + hi * 8]);
      c = __builtin_amdgcn_wmma_f32_16x16x32_f16(false, a0, false, b0, (short)0, c, false, false);
      c = __builtin_amdgcn_wmma_f32_16x16x32_f16(false, a1, false, b1, (short)0, c, false, false);
      if (leftU || rightU) {            // uniform rel bucket: cheap path
#pragma unroll
        for (int r = 0; r < 8; ++r) {
          float z = c[r] * 0.125f + (rightU ? bz6[r] : bz0[r]);
          float nm = fmaxf(mr[r], z);
          lr[r] = lr[r] * __expf(mr[r] - nm) + __expf(z - nm);
          mr[r] = nm;
        }
      } else {                          // diagonal tile: per-element bucket
        int kg = kt + t * 16 + ln;
#pragma unroll
        for (int r = 0; r < 8; ++r) {
          int ql = wave * 16 + hi * 8 + r;
          int rel = kg - (q0 + ql);
          int bidx = rel < -3 ? 0 : (rel > 3 ? 6 : rel + 3);
          float z = (c[r] + qr_s[ql][bidx]) * 0.125f;
          float nm = fmaxf(mr[r], z);
          lr[r] = lr[r] * __expf(mr[r] - nm) + __expf(z - nm);
          mr[r] = nm;
        }
      }
    }
  }
  // combine (m,l) across the 16 lanes sharing each C-row
#pragma unroll
  for (int r = 0; r < 8; ++r) {
    float mm = mr[r], ll = lr[r];
#pragma unroll
    for (int off = 1; off < 16; off <<= 1) {
      float mo = __shfl_xor(mm, off, 32);
      float lo = __shfl_xor(ll, off, 32);
      float nm = fmaxf(mm, mo);
      ll = ll * __expf(mm - nm) + lo * __expf(mo - nm);
      mm = nm;
    }
    mr[r] = mm;
    lr[r] = 1.f / ll;
  }

  // ---------------- pass 2: probs, PV, bucket masses ----------------
  v8f o[4] = {};
  float shi[8];
#pragma unroll
  for (int r = 0; r < 8; ++r) shi[r] = 0.f;

  __syncthreads();   // pass-1 consumers done before re-priming buffer 0
#if HAVE_TDM
  if (tid < 32)
    tdm_load_2d_f16<HD, KT, HID>((unsigned)(size_t)&Ks[0][0][0], kh + base);
#endif
  for (int kt = 0; kt < SEQ; kt += KT) {
    const int cur = (kt / KT) & 1;
    __syncthreads();
#if HAVE_TDM
    if (tid < 32) {
      if (kt + KT < SEQ)
        tdm_load_2d_f16<HD, KT, HID>((unsigned)(size_t)&Ks[cur ^ 1][0][0],
                                     kh + base + (size_t)(kt + KT) * HID);
    }
#endif
    for (int i = tid; i < KT * HD / 8; i += 256) {  // V tile, transposed
      int key = i >> 3, dc = (i & 7) << 3;
#if !HAVE_TDM
      *(v8h*)&Ks[cur][key][dc] =
          *(const v8h*)(kh + base + (size_t)(kt + key) * HID + dc);
#endif
      v8h vv = *(const v8h*)(vh + base + (size_t)(kt + key) * HID + dc);
#pragma unroll
      for (int j = 0; j < 8; ++j) Vt[dc + j][key] = vv[j];
    }
    if (kt + KT < SEQ)
      __builtin_prefetch(vh + base + (size_t)(kt + KT + (tid >> 2)) * HID +
                             (size_t)(tid & 3) * 16, 0, 0);
#if HAVE_TDM
    if (tid < 32) {
      if (kt + KT < SEQ) __builtin_amdgcn_s_wait_tensorcnt(1);
      else               __builtin_amdgcn_s_wait_tensorcnt(0);
    }
#endif
    __syncthreads();

    const bool leftU  = (kt + KT - 1) <= (qw0 - 4);
    const bool rightU = kt >= (qw0 + 18);
#pragma unroll
    for (int t = 0; t < 4; ++t) {
      v8f c = {};
      v16h b0 = join8(*(const v8h*)&Ks[cur][t * 16 + ln][hi * 8],
                      *(const v8h*)&Ks[cur][t * 16 + ln][16 + hi * 8]);
      v16h b1 = join8(*(const v8h*)&Ks[cur][t * 16 + ln][32 + hi * 8],
                      *(const v8h*)&Ks[cur][t * 16 + ln][48 + hi * 8]);
      c = __builtin_amdgcn_wmma_f32_16x16x32_f16(false, a0, false, b0, (short)0, c, false, false);
      c = __builtin_amdgcn_wmma_f32_16x16x32_f16(false, a1, false, b1, (short)0, c, false, false);
      if (leftU || rightU) {
#pragma unroll
        for (int r = 0; r < 8; ++r) {
          float p = __expf(c[r] * 0.125f + (rightU ? bz6[r] : bz0[r]) - mr[r]) * lr[r];
          if (rightU) shi[r] += p;      // right-tail mass (uniform branch)
          Ps[wave][hi * 8 + r][t * 16 + ln] = (_Float16)p;
        }
      } else {
        int kg = kt + t * 16 + ln;
#pragma unroll
        for (int r = 0; r < 8; ++r) {
          int ql = wave * 16 + hi * 8 + r;
          int rel = kg - (q0 + ql);
          int bidx = rel < -3 ? 0 : (rel > 3 ? 6 : rel + 3);
          float z = (c[r] + qr_s[ql][bidx]) * 0.125f;
          float p = __expf(z - mr[r]) * lr[r];
          if (rel >= 3) shi[r] += p;
          else if (rel >= -2) cen_s[ql][rel + 2] = p;   // unique center key
          Ps[wave][hi * 8 + r][t * 16 + ln] = (_Float16)p;
        }
      }
    }
    __syncthreads();

    const _Float16* prow = &Ps[wave][ln][0];
    v16h p0 = join8(*(const v8h*)(prow + hi * 8),
                    *(const v8h*)(prow + 16 + hi * 8));
    v16h p1 = join8(*(const v8h*)(prow + 32 + hi * 8),
                    *(const v8h*)(prow + 48 + hi * 8));
#pragma unroll
    for (int t = 0; t < 4; ++t) {
      v16h vb0 = join8(*(const v8h*)&Vt[t * 16 + ln][hi * 8],
                       *(const v8h*)&Vt[t * 16 + ln][16 + hi * 8]);
      v16h vb1 = join8(*(const v8h*)&Vt[t * 16 + ln][32 + hi * 8],
                       *(const v8h*)&Vt[t * 16 + ln][48 + hi * 8]);
      o[t] = __builtin_amdgcn_wmma_f32_16x16x32_f16(false, p0, false, vb0, (short)0, o[t], false, false);
      o[t] = __builtin_amdgcn_wmma_f32_16x16x32_f16(false, p1, false, vb1, (short)0, o[t], false, false);
    }
  }

#pragma unroll
  for (int r = 0; r < 8; ++r) {
#pragma unroll
    for (int off = 1; off < 16; off <<= 1) shi[r] += __shfl_xor(shi[r], off, 32);
  }
  __syncthreads();

#pragma unroll
  for (int r = 0; r < 8; ++r) {
    int ql = wave * 16 + hi * 8 + r;
    int qg = q0 + ql;
    float c0 = cen_s[ql][0], c1 = cen_s[ql][1], c2 = cen_s[ql][2],
          c3 = cen_s[ql][3], c4 = cen_s[ql][4];
    float slo = 1.f - shi[r] - (c0 + c1 + c2 + c3 + c4);
    _Float16* orow = ctx + base + (size_t)qg * HID;
#pragma unroll
    for (int t = 0; t < 4; ++t) {
      int d = t * 16 + ln;
      float w2 = slo * relv[d] + shi[r] * relv[6 * HD + d] +
                 c0 * relv[HD + d] + c1 * relv[2 * HD + d] +
                 c2 * relv[3 * HD + d] + c3 * relv[4 * HD + d] +
                 c4 * relv[5 * HD + d];
      orow[d] = (_Float16)(o[t][r] + w2);
    }
  }
}

// ---------------------------------------------------------------------------
extern "C" void kernel_launch(void* const* d_in, const int* in_sizes, int n_in,
                              void* d_out, int out_size, void* d_ws,
                              size_t ws_size, hipStream_t stream) {
  (void)in_sizes; (void)n_in; (void)out_size; (void)ws_size;
  const float* query = (const float*)d_in[0];
  const float* key_  = (const float*)d_in[1];
  const float* value = (const float*)d_in[2];
  const float* Wq = (const float*)d_in[3];  const float* bq = (const float*)d_in[4];
  const float* Wk = (const float*)d_in[5];  const float* bk = (const float*)d_in[6];
  const float* Wv = (const float*)d_in[7];  const float* bv = (const float*)d_in[8];
  const float* Wo = (const float*)d_in[9];  const float* bo = (const float*)d_in[10];
  const float* relk = (const float*)d_in[11];
  const float* relv = (const float*)d_in[12];

  const size_t nEl = (size_t)BATCH * SEQ * HID;
  _Float16* qbf = (_Float16*)d_ws;
  _Float16* kbf = qbf + nEl;
  _Float16* vbf = kbf + nEl;
  _Float16* ctx = vbf + nEl;

  dim3 gg((BATCH * SEQ / 64) * (HID / 64));  // 1024 blocks
  gemm_rowmajor_wt<float><<<gg, 128, 0, stream>>>(query, Wq, bq, nullptr, qbf);
  gemm_rowmajor_wt<float><<<gg, 128, 0, stream>>>(key_,  Wk, bk, nullptr, kbf);
  gemm_rowmajor_wt<float><<<gg, 128, 0, stream>>>(value, Wv, bv, nullptr, vbf);

  relattn_kernel<<<dim3(BATCH * NHEADS * (SEQ / 128)), 256, 0, stream>>>(
      qbf, kbf, vbf, relk, relv, ctx);

  gemm_rowmajor_wt<_Float16><<<gg, 128, 0, stream>>>(ctx, Wo, bo,
                                                     (float*)d_out, nullptr);
}
